// ValuesCoordinatesAttentionInternal_32581621908168
// MI455X (gfx1250) — compile-verified, matrix operand order checked by reading, not run on hardware
//
#include <hip/hip_runtime.h>
#include <hip/hip_bf16.h>

// ---------------------------------------------------------------------------
// Problem constants
// ---------------------------------------------------------------------------
static constexpr int BATCH   = 4;
static constexpr int SEQ     = 2048;
static constexpr int VDIM    = 1024;   // VALUES_DIM
static constexpr int CDIM    = 128;    // COORDS_DIM
static constexpr int IDIM    = 1024;   // INNER_DIM
static constexpr int HEADS   = 16;
static constexpr int DH      = 64;     // DIM_HEAD
static constexpr float EPSV  = 1e-6f;
static constexpr float SCALE = 0.125f; // 64^-0.5

// ---------------------------------------------------------------------------
// Types
// ---------------------------------------------------------------------------
typedef __bf16 bf16_t;
typedef bf16_t v16bf __attribute__((ext_vector_type(16)));
typedef float  v8f   __attribute__((ext_vector_type(8)));
typedef unsigned int v8u  __attribute__((ext_vector_type(8)));
typedef unsigned int u32x4 __attribute__((ext_vector_type(4)));
typedef int          i32x8 __attribute__((ext_vector_type(8)));
typedef int          i32x4 __attribute__((ext_vector_type(4)));

__device__ inline v8f vzero8() {
  v8f v;
  for (int i = 0; i < 8; ++i) v[i] = 0.0f;
  return v;
}

// fp32 -> bf16 (round-to-nearest-even) raw bits
__device__ inline unsigned short f2bf(float f) {
  unsigned int u = __builtin_bit_cast(unsigned int, f);
  unsigned int r = u + 0x7FFFu + ((u >> 16) & 1u);
  return (unsigned short)(r >> 16);
}

// 16x32 bf16 fragment: two 16B chunks at p and p+off2 (shorts). LDS or global.
__device__ inline v16bf ld_frag(const unsigned short* p, int off2) {
  uint4 x = *(const uint4*)(p);
  uint4 y = *(const uint4*)(p + off2);
  v8u t;
  t[0] = x.x; t[1] = x.y; t[2] = x.z; t[3] = x.w;
  t[4] = y.x; t[5] = y.y; t[6] = y.z; t[7] = y.w;
  return __builtin_bit_cast(v16bf, t);
}

__device__ inline v8f wmma_bf16(v16bf a, v16bf b, v8f c) {
  return __builtin_amdgcn_wmma_f32_16x16x32_bf16(false, a, false, b,
                                                 (short)0, c, false, false);
}

// Low 32 bits of a generic pointer to LDS == LDS byte offset (aperture rule).
__device__ inline unsigned lds_off(const void* p) {
  return (unsigned)(unsigned long long)(uintptr_t)p;
}

// ---------------------------------------------------------------------------
// TDM: DMA a 2D bf16 tile [tile1 rows x 32 cols] from global (row stride =
// strideElems bf16) into LDS at lds_byte_off, padding each 64B row by 16B
// (matching the 40-short LDS row stride used by the fragment loader).
// D# layout per CDNA5 ISA ch.8.  Issued by one wave; tracked by TENSORcnt.
// This toolchain exposes the 6-arg builtin:
//   (u32x4 g0, i32x8 g1, i32x4 g2, i32x4 g3, i32x8 g4, int cpol)
// ---------------------------------------------------------------------------
__device__ inline void tdm_load_tile(unsigned lds_byte_off, const void* gptr,
                                     int strideElems, int tile1) {
  unsigned long long ga = (unsigned long long)(uintptr_t)gptr;
  u32x4 g0;
  g0[0] = 1u;                                   // count=1, user descriptor
  g0[1] = lds_byte_off;                         // lds_addr
  g0[2] = (unsigned)(ga & 0xffffffffu);         // global_addr[31:0]
  g0[3] = (unsigned)((ga >> 32) & 0x01ffffffu)  // global_addr[56:32]
        | (2u << 30);                           // type=2 ("image")
  i32x8 g1;
  g1[0] = (int)((1u << 16)      // data_size = 2 bytes
              | (1u << 20)      // pad_enable
              | (3u << 22)      // pad_interval: every 16 DWORDs (one 64B row)
              | (3u << 25));    // pad_amount: 4 DWORDs (16B)
  g1[1] = (int)(32u << 16);                     // tensor_dim0[15:0]=32
  g1[2] = (int)(((unsigned)tile1 & 0xffffu) << 16); // tensor_dim1[15:0]
  g1[3] = (int)(32u << 16);                     // tile_dim0 = 32
  g1[4] = (int)((unsigned)tile1 & 0xffffu);     // tile_dim1; tile_dim2=0
  g1[5] = strideElems;                          // tensor_dim0_stride[31:0]
  g1[6] = 0;
  g1[7] = 0;
  i32x4 g2 = {0, 0, 0, 0};
  i32x4 g3 = {0, 0, 0, 0};
  i32x8 g4 = {0, 0, 0, 0, 0, 0, 0, 0};
  __builtin_amdgcn_tensor_load_to_lds(g0, g1, g2, g3, g4, 0);
}

// ---------------------------------------------------------------------------
// bf16-in / fp32-out GEMM via WMMA with TDM double-buffered LDS staging.
// C[M,N] = A[M,K] * Bt[N,K]^T (+bias).  M%128==0, N%128==0, K%32==0.
// 256 threads (8 waves); block tile 128x128, K-step 32; wave tile 32x64.
// ---------------------------------------------------------------------------
static constexpr int GBM = 128, GBN = 128, GBK = 32;
static constexpr int TSTR = 40;  // padded LDS row stride in shorts (64B + 16B)

__global__ __launch_bounds__(256) void gemm_tdm_kernel(
    const unsigned short* __restrict__ A, const unsigned short* __restrict__ Bt,
    const float* __restrict__ bias, float* __restrict__ C,
    int M, int N, int K) {
  (void)M;
  __shared__ __attribute__((aligned(16))) unsigned short As[2][GBM * TSTR];
  __shared__ __attribute__((aligned(16))) unsigned short Bs[2][GBN * TSTR];

  const int tid   = threadIdx.x;
  const int lane  = tid & 31;
  const int wave  = tid >> 5;
  const int waveR = wave & 3;   // 0..3
  const int waveC = wave >> 2;  // 0..1
  const int bm    = blockIdx.y * GBM;
  const int bn    = blockIdx.x * GBN;
  const int lm    = lane & 15;
  const int kh    = (lane >> 4) * 8;
  const bool issuer = (wave == 0);

  v8f acc[2][4];
  for (int i = 0; i < 2; ++i)
    for (int j = 0; j < 4; ++j) acc[i][j] = vzero8();

  const unsigned short* Abase = A  + (size_t)bm * K;
  const unsigned short* Bbase = Bt + (size_t)bn * K;

  if (issuer) {
    tdm_load_tile(lds_off(As[0]), Abase, K, GBM);
    tdm_load_tile(lds_off(Bs[0]), Bbase, K, GBN);
    __builtin_amdgcn_s_wait_tensorcnt(0);
  }
  __syncthreads();

  const int nsteps = K / GBK;
  for (int s = 0; s < nsteps; ++s) {
    const int cur = s & 1, nxt = cur ^ 1;
    if (issuer && (s + 1) < nsteps) {   // prefetch next tiles while computing
      const int k0 = (s + 1) * GBK;
      tdm_load_tile(lds_off(As[nxt]), Abase + k0, K, GBM);
      tdm_load_tile(lds_off(Bs[nxt]), Bbase + k0, K, GBN);
    }

    v16bf a[2], b[4];
    for (int mr = 0; mr < 2; ++mr)
      a[mr] = ld_frag(As[cur] + (waveR * 32 + mr * 16 + lm) * TSTR + kh, 16);
    for (int nc = 0; nc < 4; ++nc)
      b[nc] = ld_frag(Bs[cur] + (waveC * 64 + nc * 16 + lm) * TSTR + kh, 16);

    for (int mr = 0; mr < 2; ++mr)
      for (int nc = 0; nc < 4; ++nc)
        acc[mr][nc] = wmma_bf16(a[mr], b[nc], acc[mr][nc]);

    if (issuer) __builtin_amdgcn_s_wait_tensorcnt(0);
    __syncthreads();
  }

  // C-layout store: lane<16 -> M=r, lane>=16 -> M=8+r
  const int rofs = (lane < 16) ? 0 : 8;
  for (int mr = 0; mr < 2; ++mr)
    for (int nc = 0; nc < 4; ++nc) {
      int col   = bn + waveC * 64 + nc * 16 + lm;
      float bv  = bias ? bias[col] : 0.0f;
      int rbase = bm + waveR * 32 + mr * 16 + rofs;
      for (int r = 0; r < 8; ++r)
        C[(size_t)(rbase + r) * N + col] = acc[mr][nc][r] + bv;
    }
}

// ---------------------------------------------------------------------------
// fp32 -> bf16 bulk convert (n % 1024 == 0)
// ---------------------------------------------------------------------------
__global__ __launch_bounds__(256) void conv_bf16_kernel(
    const float* __restrict__ src, unsigned short* __restrict__ dst) {
  size_t i = ((size_t)blockIdx.x * 256 + threadIdx.x) * 4;
  float4 f = *(const float4*)(src + i);
  dst[i + 0] = f2bf(f.x); dst[i + 1] = f2bf(f.y);
  dst[i + 2] = f2bf(f.z); dst[i + 3] = f2bf(f.w);
}

// ---------------------------------------------------------------------------
// W[K,N] fp32 -> Wt[N,K] bf16 (32x32 tiles via LDS). K%32==0, N%32==0.
// ---------------------------------------------------------------------------
__global__ __launch_bounds__(256) void transpose_bf16_kernel(
    const float* __restrict__ W, unsigned short* __restrict__ Wt,
    int K, int N) {
  __shared__ float t[32][33];
  const int k0 = blockIdx.y * 32, n0 = blockIdx.x * 32;
  const int tx = threadIdx.x & 31, ty = threadIdx.x >> 5;
  for (int i = 0; i < 4; ++i)
    t[ty + i * 8][tx] = W[(size_t)(k0 + ty + i * 8) * N + n0 + tx];
  __syncthreads();
  for (int i = 0; i < 4; ++i)
    Wt[(size_t)(n0 + ty + i * 8) * K + k0 + tx] = f2bf(t[tx][ty + i * 8]);
}

// ---------------------------------------------------------------------------
// Block reduction helper (256 threads, wave32)
// ---------------------------------------------------------------------------
__device__ inline float block_reduce_sum(float v, float* sm) {
  v += __shfl_xor(v, 1);  v += __shfl_xor(v, 2);  v += __shfl_xor(v, 4);
  v += __shfl_xor(v, 8);  v += __shfl_xor(v, 16);
  const int wv = threadIdx.x >> 5, ln = threadIdx.x & 31;
  __syncthreads();
  if (ln == 0) sm[wv] = v;
  __syncthreads();
  float t = 0.0f;
  for (int i = 0; i < 8; ++i) t += sm[i];
  return t;
}

// ---------------------------------------------------------------------------
// RMS-norm qv/kv over 1024 + reshape to bf16 [B,H,N,DH]; vv -> [B,H,DH,N]
// ---------------------------------------------------------------------------
__global__ __launch_bounds__(256) void qkvnorm_kernel(
    const float* __restrict__ qkv, const float* __restrict__ qw,
    const float* __restrict__ kw, unsigned short* __restrict__ Q,
    unsigned short* __restrict__ Kv, unsigned short* __restrict__ Vt) {
  __shared__ float sm[8];
  const int row = blockIdx.x;
  const int b   = row >> 11;
  const int n   = row & (SEQ - 1);
  const int tid = threadIdx.x;
  const float* base = qkv + (size_t)row * (3 * IDIM);

  float v[4], ss = 0.0f;
  for (int i = 0; i < 4; ++i) { v[i] = base[tid + i * 256]; ss += v[i] * v[i]; }
  float r = rsqrtf(block_reduce_sum(ss, sm) * (1.0f / IDIM) + EPSV);
  for (int i = 0; i < 4; ++i) {
    int e = tid + i * 256;
    Q[(((size_t)b * HEADS + (e >> 6)) * SEQ + n) * DH + (e & 63)] =
        f2bf(v[i] * r * qw[e]);
  }
  ss = 0.0f;
  for (int i = 0; i < 4; ++i) { v[i] = base[IDIM + tid + i * 256]; ss += v[i] * v[i]; }
  r = rsqrtf(block_reduce_sum(ss, sm) * (1.0f / IDIM) + EPSV);
  for (int i = 0; i < 4; ++i) {
    int e = tid + i * 256;
    Kv[(((size_t)b * HEADS + (e >> 6)) * SEQ + n) * DH + (e & 63)] =
        f2bf(v[i] * r * kw[e]);
  }
  for (int i = 0; i < 4; ++i) {
    int e = tid + i * 256;
    Vt[(((size_t)b * HEADS + (e >> 6)) * DH + (e & 63)) * SEQ + n] =
        f2bf(base[2 * IDIM + e]);
  }
}

// ---------------------------------------------------------------------------
// RMS-norm coords projection over 2048 -> qc/kc bf16 [B,H,N,DH]
// ---------------------------------------------------------------------------
__global__ __launch_bounds__(256) void cnorm_kernel(
    const float* __restrict__ qkc, const float* __restrict__ cw,
    unsigned short* __restrict__ Qc, unsigned short* __restrict__ Kc) {
  __shared__ float sm[8];
  const int row = blockIdx.x;
  const int b   = row >> 11;
  const int n   = row & (SEQ - 1);
  const float* base = qkc + (size_t)row * (2 * IDIM);
  float v[8], ss = 0.0f;
  for (int i = 0; i < 8; ++i) {
    v[i] = base[threadIdx.x + i * 256];
    ss += v[i] * v[i];
  }
  float r = rsqrtf(block_reduce_sum(ss, sm) * (1.0f / (2 * IDIM)) + EPSV);
  for (int i = 0; i < 8; ++i) {
    int e = threadIdx.x + i * 256;
    unsigned short bv = f2bf(v[i] * r * cw[e]);
    int e2 = e & (IDIM - 1);
    size_t dst = (((size_t)b * HEADS + (e2 >> 6)) * SEQ + n) * DH + (e2 & 63);
    if (e < IDIM) Qc[dst] = bv; else Kc[dst] = bv;
  }
}

// ---------------------------------------------------------------------------
// Flash attention. Grid (SEQ/128, B*H), 256 threads = 8 waves.
// Each wave owns 16 query rows; online softmax over key blocks of 32.
// Output written as bf16 [B, N, IDIM] for the TDM final GEMM.
// ---------------------------------------------------------------------------
__global__ __launch_bounds__(256) void attn_kernel(
    const unsigned short* __restrict__ Q, const unsigned short* __restrict__ Kv,
    const unsigned short* __restrict__ Qc, const unsigned short* __restrict__ Kc,
    const unsigned short* __restrict__ Vt, const unsigned char* __restrict__ mask,
    const float* __restrict__ relw, unsigned short* __restrict__ Out) {
  __shared__ __attribute__((aligned(16))) unsigned short Pt[8][16 * 32];

  const int tid  = threadIdx.x;
  const int lane = tid & 31;
  const int wave = tid >> 5;
  const int bh   = blockIdx.y;
  const int b    = bh >> 4;
  const int h    = bh & (HEADS - 1);
  const int q0   = blockIdx.x * 128 + wave * 16;
  const float w  = relw[0];
  const int lm   = lane & 15;
  const int kh   = (lane >> 4) * 8;

  v16bf aq[2], ac[2];
  {
    size_t base = ((size_t)bh * SEQ + q0 + lm) * DH;
    for (int c = 0; c < 2; ++c) {
      aq[c] = ld_frag(Q  + base + c * 32 + kh, 16);
      ac[c] = ld_frag(Qc + base + c * 32 + kh, 16);
    }
  }

  v8f O[4];
  for (int j = 0; j < 4; ++j) O[j] = vzero8();
  float mi[8], li[8];
  for (int r = 0; r < 8; ++r) { mi[r] = -INFINITY; li[r] = 0.0f; }

  for (int kb = 0; kb < SEQ; kb += 32) {
    v16bf bk[2][2], bc[2][2];
    for (int sub = 0; sub < 2; ++sub) {
      size_t base = ((size_t)bh * SEQ + kb + sub * 16 + lm) * DH;
      for (int c = 0; c < 2; ++c) {
        bk[sub][c] = ld_frag(Kv + base + c * 32 + kh, 16);
        bc[sub][c] = ld_frag(Kc + base + c * 32 + kh, 16);
      }
    }
    v8f s[2];
    for (int sub = 0; sub < 2; ++sub) {
      v8f sv = vzero8(), sc = vzero8();
      sv = wmma_bf16(aq[0], bk[sub][0], sv);
      sv = wmma_bf16(aq[1], bk[sub][1], sv);
      sc = wmma_bf16(ac[0], bc[sub][0], sc);
      sc = wmma_bf16(ac[1], bc[sub][1], sc);
      unsigned char mk = mask[(size_t)b * SEQ + kb + sub * 16 + lm];
      for (int r = 0; r < 8; ++r)
        s[sub][r] = mk ? SCALE * (sv[r] + w * sc[r]) : -INFINITY;
    }
    float alpha[8];
    for (int r = 0; r < 8; ++r) {
      float rm = fmaxf(s[0][r], s[1][r]);
      rm = fmaxf(rm, __shfl_xor(rm, 1));
      rm = fmaxf(rm, __shfl_xor(rm, 2));
      rm = fmaxf(rm, __shfl_xor(rm, 4));
      rm = fmaxf(rm, __shfl_xor(rm, 8));
      float mo = mi[r];
      float mn = fmaxf(mo, rm);
      float al = (mn == -INFINITY) ? 1.0f : __expf(mo - mn);
      float rs = 0.0f;
      for (int sub = 0; sub < 2; ++sub) {
        float p = (mn == -INFINITY) ? 0.0f : __expf(s[sub][r] - mn);
        s[sub][r] = p;
        rs += p;
      }
      rs += __shfl_xor(rs, 1); rs += __shfl_xor(rs, 2);
      rs += __shfl_xor(rs, 4); rs += __shfl_xor(rs, 8);
      li[r]    = li[r] * al + rs;
      mi[r]    = mn;
      alpha[r] = al;
    }
    for (int j = 0; j < 4; ++j)
      for (int r = 0; r < 8; ++r) O[j][r] *= alpha[r];

    unsigned short* pt = Pt[wave];
    {
      const int rofs = (lane < 16) ? 0 : 8;
      for (int sub = 0; sub < 2; ++sub)
        for (int r = 0; r < 8; ++r)
          pt[(rofs + r) * 32 + sub * 16 + lm] = f2bf(s[sub][r]);
    }
    asm volatile("s_wait_dscnt 0" ::: "memory");
    v16bf aP = ld_frag(pt + lm * 32 + kh, 16);

    for (int j = 0; j < 4; ++j) {
      size_t base = ((size_t)bh * DH + j * 16 + lm) * SEQ + kb + kh;
      v16bf bv = ld_frag(Vt + base, 16);
      O[j] = wmma_bf16(aP, bv, O[j]);
    }
  }

  const int rofs = (lane < 16) ? 0 : 8;
  for (int r = 0; r < 8; ++r) {
    float inv = (li[r] > 0.0f) ? (1.0f / li[r]) : 0.0f;
    int row = q0 + rofs + r;
    for (int j = 0; j < 4; ++j)
      Out[((size_t)b * SEQ + row) * IDIM + h * DH + j * 16 + lm] =
          f2bf(O[j][r] * inv);
  }
}

// ---------------------------------------------------------------------------
// Host-side launch
// ---------------------------------------------------------------------------
extern "C" void kernel_launch(void* const* d_in, const int* in_sizes, int n_in,
                              void* d_out, int out_size, void* d_ws, size_t ws_size,
                              hipStream_t stream) {
  (void)in_sizes; (void)n_in; (void)out_size; (void)ws_size;
  const float*         values = (const float*)d_in[0];
  const float*         coords = (const float*)d_in[1];
  const unsigned char* amask  = (const unsigned char*)d_in[2];
  const float*         Wqkv   = (const float*)d_in[3];
  const float*         qw     = (const float*)d_in[4];
  const float*         kw     = (const float*)d_in[5];
  const float*         Wqkc   = (const float*)d_in[6];
  const float*         cw     = (const float*)d_in[7];
  const float*         relw   = (const float*)d_in[8];
  const float*         Wout   = (const float*)d_in[9];
  const float*         bout   = (const float*)d_in[10];
  float*               out    = (float*)d_out;

  const size_t ROWS = (size_t)BATCH * SEQ;  // 8192
  char* ws = (char*)d_ws;
  size_t off = 0;
  float* qkv = (float*)(ws + off);           off += ROWS * 3 * IDIM * sizeof(float); // 96 MB
  float* qkc = (float*)(ws + off);           off += ROWS * 2 * IDIM * sizeof(float); // 64 MB
  unsigned short* Qb   = (unsigned short*)(ws + off); off += ROWS * IDIM * 2;        // 16 MB
  unsigned short* Kb   = (unsigned short*)(ws + off); off += ROWS * IDIM * 2;
  unsigned short* Vtb  = (unsigned short*)(ws + off); off += ROWS * IDIM * 2;
  unsigned short* Qcb  = (unsigned short*)(ws + off); off += ROWS * IDIM * 2;
  unsigned short* Kcb  = (unsigned short*)(ws + off); off += ROWS * IDIM * 2;
  unsigned short* attb = (unsigned short*)(ws + off); off += ROWS * IDIM * 2;        // 16 MB
  unsigned short* valb = (unsigned short*)(ws + off); off += ROWS * VDIM * 2;        // 16 MB
  unsigned short* crdb = (unsigned short*)(ws + off); off += ROWS * CDIM * 2;        //  2 MB
  unsigned short* WqkvT = (unsigned short*)(ws + off); off += (size_t)3 * IDIM * VDIM * 2;
  unsigned short* WqkcT = (unsigned short*)(ws + off); off += (size_t)2 * IDIM * CDIM * 2;
  unsigned short* WoutT = (unsigned short*)(ws + off); off += (size_t)VDIM * IDIM * 2;

  // 0) bf16 conversions + weight transposes (one-time, bandwidth-cheap)
  conv_bf16_kernel<<<dim3((unsigned)(ROWS * VDIM / 1024)), 256, 0, stream>>>(values, valb);
  conv_bf16_kernel<<<dim3((unsigned)(ROWS * CDIM / 1024)), 256, 0, stream>>>(coords, crdb);
  transpose_bf16_kernel<<<dim3(3 * IDIM / 32, VDIM / 32), 256, 0, stream>>>(Wqkv, WqkvT, VDIM, 3 * IDIM);
  transpose_bf16_kernel<<<dim3(2 * IDIM / 32, CDIM / 32), 256, 0, stream>>>(Wqkc, WqkcT, CDIM, 2 * IDIM);
  transpose_bf16_kernel<<<dim3(VDIM / 32, IDIM / 32), 256, 0, stream>>>(Wout, WoutT, IDIM, VDIM);

  // 1) qkv = values @ Wqkv   (TDM-staged WMMA GEMM)
  gemm_tdm_kernel<<<dim3(3 * IDIM / GBN, ROWS / GBM), 256, 0, stream>>>(
      valb, WqkvT, nullptr, qkv, (int)ROWS, 3 * IDIM, VDIM);
  // 2) rms-norm qv/kv + head reshape, vv transposed
  qkvnorm_kernel<<<dim3((unsigned)ROWS), 256, 0, stream>>>(qkv, qw, kw, Qb, Kb, Vtb);
  // 3) qkc = coords @ Wqk_c
  gemm_tdm_kernel<<<dim3(2 * IDIM / GBN, ROWS / GBM), 256, 0, stream>>>(
      crdb, WqkcT, nullptr, qkc, (int)ROWS, 2 * IDIM, CDIM);
  // 4) rms-norm over 2048 -> qc/kc
  cnorm_kernel<<<dim3((unsigned)ROWS), 256, 0, stream>>>(qkc, cw, Qcb, Kcb);
  // 5) flash attention -> bf16
  attn_kernel<<<dim3(SEQ / 128, BATCH * HEADS), 256, 0, stream>>>(
      Qb, Kb, Qcb, Kcb, Vtb, amask, relw, attb);
  // 6) out = attnout @ Wout + bout
  gemm_tdm_kernel<<<dim3(VDIM / GBN, ROWS / GBM), 256, 0, stream>>>(
      attb, WoutT, bout, out, (int)ROWS, VDIM, IDIM);
}
